// LowrankLearnableHash_42434276884848
// MI455X (gfx1250) — matrix operand order repaired and checked by reading, not compiled, not run on hardware
//
#include <hip/hip_runtime.h>

typedef __attribute__((ext_vector_type(16))) _Float16 v16h;
typedef __attribute__((ext_vector_type(8)))  _Float16 v8h;
typedef __attribute__((ext_vector_type(8)))  float    v8f;

#define NRAYS   1024
#define NSAMP   512
#define RPLANE  512
#define TPB     128
#define RADIUS_F 1.3f
#define STEP_F  (2.0f * RADIUS_F / 512.0f)

static __device__ __forceinline__ int imin(int a, int b) { return a < b ? a : b; }

// ---- WMMA fragment helpers (f16 16x16x32, wave32 layouts per CDNA5 ISA 7.12.2) ----

// A matrix 16x32 f16: lanes 0-15 row M=lane, halves 0..7 = K 0..7, 8..15 = K 16..23;
// lanes 16-31 row M=lane-16, halves 0..7 = K 8..15, 8..15 = K 24..31.
static __device__ __forceinline__ v16h ldA(const _Float16* act, int stride, int mBase,
                                           int kBase, int lane) {
  int m  = mBase + (lane & 15);
  int hi = (lane & 16) ? 8 : 0;
  const _Float16* row = act + m * stride + kBase;
  v8h lo8 = *(const v8h*)(row + hi);
  v8h hi8 = *(const v8h*)(row + 16 + hi);
  v16h r;
#pragma unroll
  for (int i = 0; i < 8; ++i) { r[i] = lo8[i]; r[8 + i] = hi8[i]; }
  return r;
}

// B matrix 32x16 f16 from N-major LDS weights wt[n*kStride + k]:
// lanes 0-15 col N=lane, K=kBase+0..7,16..23; lanes 16-31 col N=lane-16, K=kBase+8..15,24..31.
static __device__ __forceinline__ v16h ldB(const _Float16* wt, int kStride, int nBase,
                                           int kBase, int lane) {
  int n  = nBase + (lane & 15);
  int k0 = kBase + ((lane & 16) ? 8 : 0);
  const _Float16* p = wt + n * kStride + k0;
  v8h lo8 = *(const v8h*)(p);
  v8h hi8 = *(const v8h*)(p + 16);
  v16h r;
#pragma unroll
  for (int i = 0; i < 8; ++i) { r[i] = lo8[i]; r[8 + i] = hi8[i]; }
  return r;
}

// C/D 16x16 f32: VGPR r holds M = r (lanes 0-15) / M = 8+r (lanes 16-31), N = lane%16.
static __device__ __forceinline__ void stD_act(_Float16* act, int stride, int mBase,
                                               int nBase, int lane, v8f d,
                                               const float* bias, bool relu_on) {
  int n  = nBase + (lane & 15);
  int mo = mBase + ((lane & 16) ? 8 : 0);
  float bv = bias[n];
#pragma unroll
  for (int r = 0; r < 8; ++r) {
    float v = d[r] + bv;
    if (relu_on) v = fmaxf(v, 0.0f);
    act[(mo + r) * stride + n] = (_Float16)v;
  }
}

static __device__ __forceinline__ v8f wmma32(v16h a, v16h b, v8f c) {
  return __builtin_amdgcn_wmma_f32_16x16x32_f16(false, a, false, b, (short)0, c,
                                                false, false);
}

// ---- bilinear plane sample fused with basis_W projection ----
static __device__ __forceinline__ void plane_accum(const float* __restrict__ G, float u,
                                                   float v, const float* bw, int kOff,
                                                   float gv[3]) {
  float pu = fminf(fmaxf((u + 1.0f) * 0.5f * (float)(RPLANE - 1), 0.0f),
                   (float)(RPLANE - 1));
  float pv = fminf(fmaxf((v + 1.0f) * 0.5f * (float)(RPLANE - 1), 0.0f),
                   (float)(RPLANE - 1));
  int iu = imin((int)pu, RPLANE - 2);
  int iv = imin((int)pv, RPLANE - 2);
  float fu = pu - (float)iu, fv = pv - (float)iv;
  float w00 = (1.0f - fv) * (1.0f - fu), w01 = (1.0f - fv) * fu;
  float w10 = fv * (1.0f - fu),          w11 = fv * fu;
  const float* base = G + (size_t)iv * RPLANE + iu;
#pragma unroll 4
  for (int ch = 0; ch < 16; ++ch) {
    const float* g = base + (size_t)ch * RPLANE * RPLANE;
    float val = w00 * g[0] + w01 * g[1] + w10 * g[RPLANE] + w11 * g[RPLANE + 1];
    int k = kOff + ch;
    gv[0] += val * bw[k];
    gv[1] += val * bw[48 + k];
    gv[2] += val * bw[96 + k];
  }
}

__global__ __launch_bounds__(TPB) void nerf_sample_kernel(
    const float* __restrict__ rays_o, const float* __restrict__ rays_d,
    const float* __restrict__ Gxy, const float* __restrict__ Gxz,
    const float* __restrict__ Gyz, const float* __restrict__ basis_W,
    const float* __restrict__ Fgrid,
    const float* __restrict__ sW1, const float* __restrict__ sb1,
    const float* __restrict__ sW2, const float* __restrict__ sb2,
    const float* __restrict__ cW1, const float* __restrict__ cb1,
    const float* __restrict__ cW2, const float* __restrict__ cb2,
    const float* __restrict__ cW3, const float* __restrict__ cb3,
    float* __restrict__ sigma_out, float* __restrict__ rgb_out) {
  // Weights staged N-major ([n][k]) as f16 for B-fragment vector loads.
  __shared__ __align__(16) _Float16 wS1t[64 * 32];
  __shared__ __align__(16) _Float16 wS2t[16 * 64];
  __shared__ __align__(16) _Float16 wC1t[64 * 32];
  __shared__ __align__(16) _Float16 wC2t[64 * 64];
  __shared__ __align__(16) _Float16 wC3t[16 * 64];
  __shared__ float sB1[64], sB2[16], cB1[64], cB2[64], cB3[4];
  __shared__ float basisW[144];
  __shared__ __align__(16) _Float16 actA[TPB * 32];   // feats, later cin
  __shared__ __align__(16) _Float16 actH[TPB * 64];   // hidden
  __shared__ __align__(16) char pool[TPB * 64 * 2];   // actH2 f16 / dcbuf f32 alias
  __shared__ float maskbuf[TPB];

  _Float16* actH2 = (_Float16*)pool;
  float*    dcbuf = (float*)pool;  // TPB*16 f32 = 8KB, consumed before actH2 is written

  const int t    = threadIdx.x;
  const int lane = t & 31;
  const int wave = t >> 5;

  // ---- ray setup (uniform per block: 128 | 512 -> one ray per block) ----
  const int sBase = blockIdx.x * TPB;
  const int s = sBase + t;
  const int b = s >> 9;
  const int si = s & (NSAMP - 1);

  float ox = rays_o[b * 3 + 0], oy = rays_o[b * 3 + 1], oz = rays_o[b * 3 + 2];
  float rx = rays_d[b * 3 + 0], ry = rays_d[b * 3 + 1], rz = rays_d[b * 3 + 2];
  float inv = rsqrtf(rx * rx + ry * ry + rz * rz);
  float dnx = rx * inv, dny = ry * inv, dnz = rz * inv;
  float dsx = (fabsf(dnx) < 1e-9f) ? 1e-9f : dnx;
  float dsy = (fabsf(dny) < 1e-9f) ? 1e-9f : dny;
  float dsz = (fabsf(dnz) < 1e-9f) ? 1e-9f : dnz;
  float t1x = (-RADIUS_F - ox) / dsx, t2x = (RADIUS_F - ox) / dsx;
  float t1y = (-RADIUS_F - oy) / dsy, t2y = (RADIUS_F - oy) / dsy;
  float t1z = (-RADIUS_F - oz) / dsz, t2z = (RADIUS_F - oz) / dsz;
  float tmin = fmaxf(fmaxf(fminf(t1x, t2x), fminf(t1y, t2y)), fminf(t1z, t2z));
  float tmax = fminf(fminf(fmaxf(t1x, t2x), fmaxf(t1y, t2y)), fmaxf(t1z, t2z));
  float nearv = fmaxf(tmin, 0.0f) + 1e-4f;

  // Uniform early-exit: ts is monotonic in sample index, so the whole block is
  // dead iff the block's FIRST sample is masked out (uniform condition).
  {
    float ts0 = nearv + STEP_F * (float)(sBase & (NSAMP - 1));
    bool blockAlive = (ts0 < tmax) && (tmax > nearv);
    if (!blockAlive) {
      sigma_out[s] = 0.0f;
      rgb_out[(size_t)s * 3 + 0] = 0.0f;
      rgb_out[(size_t)s * 3 + 1] = 0.0f;
      rgb_out[(size_t)s * 3 + 2] = 0.0f;
      return;
    }
  }

  // ---- stage weights / biases ----
  for (int idx = t; idx < 64 * 32; idx += TPB) {
    int n = idx >> 5, k = idx & 31;
    wS1t[idx] = (_Float16)sW1[k * 64 + n];
    wC1t[idx] = (_Float16)((k < 31) ? cW1[k * 64 + n] : 0.0f);
  }
  for (int idx = t; idx < 16 * 64; idx += TPB) {
    int n = idx >> 6, k = idx & 63;
    wS2t[idx] = (_Float16)sW2[k * 16 + n];
    wC3t[idx] = (_Float16)((n < 3) ? cW3[k * 3 + n] : 0.0f);
  }
  for (int idx = t; idx < 64 * 64; idx += TPB) {
    int n = idx >> 6, k = idx & 63;
    wC2t[idx] = (_Float16)cW2[k * 64 + n];
  }
  if (t < 64) { sB1[t] = sb1[t]; cB1[t] = cb1[t]; cB2[t] = cb2[t]; }
  if (t < 16) sB2[t] = sb2[t];
  if (t < 4)  cB3[t] = (t < 3) ? cb3[t] : 0.0f;
  for (int idx = t; idx < 144; idx += TPB) basisW[idx] = basis_W[idx];
  __syncthreads();

  // ---- phase 1: per-sample geometry + grid gathers ----
  float tsv = nearv + STEP_F * (float)si;
  float maskv = ((tsv < tmax) && (tmax > nearv)) ? 1.0f : 0.0f;
  float px = fminf(fmaxf((ox + dnx * tsv) / RADIUS_F, -1.0f), 1.0f);
  float py = fminf(fmaxf((oy + dny * tsv) / RADIUS_F, -1.0f), 1.0f);
  float pz = fminf(fmaxf((oz + dnz * tsv) / RADIUS_F, -1.0f), 1.0f);

  float gv[3] = {0.0f, 0.0f, 0.0f};
  plane_accum(Gxy, px, py, basisW, 0,  gv);   // (x->cols, y->rows)
  plane_accum(Gxz, px, pz, basisW, 16, gv);
  plane_accum(Gyz, py, pz, basisW, 32, gv);

  // trilinear in Fgrid [32,16,16,16]; coords reversed: gv[2]->dim0, gv[1]->dim1, gv[0]->dim2
  float p0 = fminf(fmaxf((gv[2] + 1.0f) * 0.5f * 15.0f, 0.0f), 15.0f);
  float p1 = fminf(fmaxf((gv[1] + 1.0f) * 0.5f * 15.0f, 0.0f), 15.0f);
  float p2 = fminf(fmaxf((gv[0] + 1.0f) * 0.5f * 15.0f, 0.0f), 15.0f);
  int i0 = imin((int)p0, 14), i1 = imin((int)p1, 14), i2 = imin((int)p2, 14);
  float f0 = p0 - (float)i0, f1 = p1 - (float)i1, f2 = p2 - (float)i2;
  float cw[8]; int coff[8];
#pragma unroll
  for (int c = 0; c < 8; ++c) {
    int d0 = c & 1, d1 = (c >> 1) & 1, d2 = (c >> 2) & 1;
    cw[c] = (d0 ? f0 : 1.0f - f0) * (d1 ? f1 : 1.0f - f1) * (d2 ? f2 : 1.0f - f2);
    coff[c] = (i0 + d0) * 256 + (i1 + d1) * 16 + (i2 + d2);
  }
#pragma unroll 4
  for (int ch = 0; ch < 32; ++ch) {
    const float* F = Fgrid + ch * 4096;
    float acc = 0.0f;
#pragma unroll
    for (int c = 0; c < 8; ++c) acc += cw[c] * F[coff[c]];
    actA[t * 32 + ch] = (_Float16)acc;
  }
  maskbuf[t] = maskv;
  __syncthreads();

  // ---- phase 2: sigma MLP (each wave owns its 2 M-tiles = its own 32 rows) ----
#pragma unroll
  for (int mi = 0; mi < 2; ++mi) {
    int mBase = (wave * 2 + mi) * 16;
    v16h aF = ldA(actA, 32, mBase, 0, lane);
#pragma unroll
    for (int nt = 0; nt < 4; ++nt) {
      v16h bF = ldB(wS1t, 32, nt * 16, 0, lane);
      v8f c = {};
      c = wmma32(aF, bF, c);
      stD_act(actH, 64, mBase, nt * 16, lane, c, sB1, true);
    }
    // dc = h @ sW2 + sb2 (no relu), K = 64
    v8f c = {};
#pragma unroll
    for (int kk = 0; kk < 64; kk += 32) {
      v16h aH = ldA(actH, 64, mBase, kk, lane);
      v16h bF = ldB(wS2t, 64, 0, kk, lane);
      c = wmma32(aH, bF, c);
    }
    {
      int n = lane & 15;
      int mo = mBase + ((lane & 16) ? 8 : 0);
      float bv = sB2[n];
#pragma unroll
      for (int r = 0; r < 8; ++r) dcbuf[(mo + r) * 16 + n] = c[r] + bv;
    }
  }
  __syncthreads();

  // ---- sigma out + cin = [sh16(dn), dc[1:16], 0] ----
  sigma_out[s] = fmaxf(dcbuf[t * 16 + 0], 0.0f) * maskv;
  {
    float x = dnx, y = dny, z = dnz;
    float xx = x * x, yy = y * y, zz = z * z;
    float xy = x * y, yz = y * z, xz = x * z;
    float sh[16];
    sh[0]  = 0.28209479177387814f;
    sh[1]  = -0.48860251190291987f * y;
    sh[2]  =  0.48860251190291987f * z;
    sh[3]  = -0.48860251190291987f * x;
    sh[4]  =  1.0925484305920792f * xy;
    sh[5]  = -1.0925484305920792f * yz;
    sh[6]  =  0.94617469575756f * zz - 0.31539156525252f;
    sh[7]  = -1.0925484305920792f * xz;
    sh[8]  =  0.5462742152960396f * (xx - yy);
    sh[9]  = -0.5900435899266435f * y * (3.0f * xx - yy);
    sh[10] =  2.890611442640554f * xy * z;
    sh[11] = -0.4570457994644657f * y * (4.0f * zz - xx - yy);
    sh[12] =  0.3731763325901154f * z * (2.0f * zz - 3.0f * xx - 3.0f * yy);
    sh[13] = -0.4570457994644657f * x * (4.0f * zz - xx - yy);
    sh[14] =  1.445305721320277f * z * (xx - yy);
    sh[15] = -0.5900435899266435f * x * (xx - 3.0f * yy);
#pragma unroll
    for (int k = 0; k < 16; ++k) actA[t * 32 + k] = (_Float16)sh[k];
#pragma unroll
    for (int k = 0; k < 15; ++k) actA[t * 32 + 16 + k] = (_Float16)dcbuf[t * 16 + 1 + k];
    actA[t * 32 + 31] = (_Float16)0.0f;
  }
  __syncthreads();

  // ---- color MLP ----
#pragma unroll
  for (int mi = 0; mi < 2; ++mi) {
    int mBase = (wave * 2 + mi) * 16;
    v16h aC = ldA(actA, 32, mBase, 0, lane);
#pragma unroll
    for (int nt = 0; nt < 4; ++nt) {
      v16h bF = ldB(wC1t, 32, nt * 16, 0, lane);
      v8f c = {};
      c = wmma32(aC, bF, c);
      stD_act(actH, 64, mBase, nt * 16, lane, c, cB1, true);
    }
#pragma unroll
    for (int nt = 0; nt < 4; ++nt) {
      v8f c = {};
#pragma unroll
      for (int kk = 0; kk < 64; kk += 32) {
        v16h aH = ldA(actH, 64, mBase, kk, lane);
        v16h bF = ldB(wC2t, 64, nt * 16, kk, lane);
        c = wmma32(aH, bF, c);
      }
      stD_act(actH2, 64, mBase, nt * 16, lane, c, cB2, true);
    }
    v8f c = {};
#pragma unroll
    for (int kk = 0; kk < 64; kk += 32) {
      v16h aH = ldA(actH2, 64, mBase, kk, lane);
      v16h bF = ldB(wC3t, 64, 0, kk, lane);
      c = wmma32(aH, bF, c);
    }
    int n = lane & 15;
    int mo = mBase + ((lane & 16) ? 8 : 0);
    if (n < 3) {
      float bv = cB3[n];
#pragma unroll
      for (int r = 0; r < 8; ++r) {
        int m = mo + r;
        float v = c[r] + bv;
        float sg = 1.0f / (1.0f + __expf(-v));
        rgb_out[(size_t)(sBase + m) * 3 + n] = sg * maskbuf[m];
      }
    }
  }
}

// ---- per-ray transmittance scan + compositing: one wave per ray ----
__global__ __launch_bounds__(32) void nerf_composite_kernel(
    const float* __restrict__ sigma, const float* __restrict__ rgb,
    float* __restrict__ out) {
  int b = blockIdx.x;
  int lane = threadIdx.x;
  float Trun = 1.0f;
  float accw = 0.0f, ar = 0.0f, ag = 0.0f, ab = 0.0f;
  for (int c0 = 0; c0 < NSAMP; c0 += 32) {
    int s = b * NSAMP + c0 + lane;
    float sg = sigma[s];
    float alpha = 1.0f - __expf(-sg * STEP_F);
    float q = 1.0f - alpha + 1e-10f;
    float p = q;
#pragma unroll
    for (int off = 1; off < 32; off <<= 1) {
      float tm = __shfl_up(p, off, 32);
      if (lane >= off) p *= tm;
    }
    float excl = __shfl_up(p, 1, 32);
    if (lane == 0) excl = 1.0f;
    float T = Trun * excl;
    float w = alpha * T;
    accw += w;
    ar += w * rgb[(size_t)s * 3 + 0];
    ag += w * rgb[(size_t)s * 3 + 1];
    ab += w * rgb[(size_t)s * 3 + 2];
    Trun *= __shfl(p, 31, 32);
  }
#pragma unroll
  for (int off = 16; off > 0; off >>= 1) {
    accw += __shfl_down(accw, off, 32);
    ar   += __shfl_down(ar,   off, 32);
    ag   += __shfl_down(ag,   off, 32);
    ab   += __shfl_down(ab,   off, 32);
  }
  if (lane == 0) {
    float bg = 1.0f - accw;
    out[b * 3 + 0] = ar + bg;
    out[b * 3 + 1] = ag + bg;
    out[b * 3 + 2] = ab + bg;
  }
}

extern "C" void kernel_launch(void* const* d_in, const int* in_sizes, int n_in,
                              void* d_out, int out_size, void* d_ws, size_t ws_size,
                              hipStream_t stream) {
  const float* rays_o  = (const float*)d_in[0];
  const float* rays_d  = (const float*)d_in[1];
  const float* Gxy     = (const float*)d_in[2];
  const float* Gxz     = (const float*)d_in[3];
  const float* Gyz     = (const float*)d_in[4];
  const float* basis_W = (const float*)d_in[5];
  const float* Fgrid   = (const float*)d_in[6];
  const float* sW1 = (const float*)d_in[7];
  const float* sb1 = (const float*)d_in[8];
  const float* sW2 = (const float*)d_in[9];
  const float* sb2 = (const float*)d_in[10];
  const float* cW1 = (const float*)d_in[11];
  const float* cb1 = (const float*)d_in[12];
  const float* cW2 = (const float*)d_in[13];
  const float* cb2 = (const float*)d_in[14];
  const float* cW3 = (const float*)d_in[15];
  const float* cb3 = (const float*)d_in[16];

  float* sigma = (float*)d_ws;
  float* rgb   = (float*)((char*)d_ws + (size_t)NRAYS * NSAMP * sizeof(float));

  dim3 g1(NRAYS * NSAMP / TPB), b1(TPB);
  nerf_sample_kernel<<<g1, b1, 0, stream>>>(rays_o, rays_d, Gxy, Gxz, Gyz, basis_W,
                                            Fgrid, sW1, sb1, sW2, sb2, cW1, cb1,
                                            cW2, cb2, cW3, cb3, sigma, rgb);
  nerf_composite_kernel<<<dim3(NRAYS), dim3(32), 0, stream>>>(sigma, rgb,
                                                              (float*)d_out);
}
